// ShardedBTLMBlock_70102456205745
// MI455X (gfx1250) — matrix operand order repaired and checked by reading, not compiled, not run on hardware
//
#include <hip/hip_runtime.h>

// ============================================================================
// ShardedBTLMBlock on gfx1250 (MI455X, wave32, WMMA f32<=f16 16x16x32).
// Compute-bound (~0.52 TFLOP/call); all GEMMs + attention on the WMMA pipe.
// - f16 operands / f32 accumulate (f32 WMMA is K=4 -> 8x less throughput).
// - Weights converted f32->f16 inside GEMM staging (no separate pass).
// - LDS B tiles stored transposed so every WMMA operand loads as ds_load_b128.
// - Pure-copy staging (GEMM A tile, attention K tile) uses async global->LDS
//   (GLOBAL_LOAD_ASYNC_TO_LDS_B128, ASYNCcnt) — builtin confirmed present.
// - Flash attention, causal+ALiBi computed analytically (~300MB HBM saved).
// ============================================================================

typedef _Float16 h16;
typedef __attribute__((ext_vector_type(16))) _Float16 v16h;
typedef __attribute__((ext_vector_type(8)))  float    v8f;
typedef int v4i __attribute__((vector_size(16)));

#define WMMA16(A, Bf, C) \
  __builtin_amdgcn_wmma_f32_16x16x32_f16(false, (A), false, (Bf), (short)0, (C), false, false)

#if defined(__has_builtin)
#if __has_builtin(__builtin_amdgcn_global_load_async_to_lds_b128)
#define HAVE_ASYNC_LDS 1
#endif
#endif

#define AS1 __attribute__((address_space(1)))
#define AS3 __attribute__((address_space(3)))

#ifdef HAVE_ASYNC_LDS
// 16B async copy global -> LDS (GLOBAL_LOAD_ASYNC_TO_LDS_B128, ASYNCcnt).
static __device__ __forceinline__ void async_copy_b128(const h16* gsrc, h16* ldst) {
  __builtin_amdgcn_global_load_async_to_lds_b128(
      (AS1 v4i*)(uintptr_t)gsrc, (AS3 v4i*)(uintptr_t)ldst, 0, 0);
}
#endif

static __device__ __forceinline__ void async_wait0() {
#if defined(__has_builtin) && __has_builtin(__builtin_amdgcn_s_wait_asynccnt)
  __builtin_amdgcn_s_wait_asynccnt(0);
#else
  asm volatile("s_wait_asynccnt 0" ::: "memory");
#endif
}

// ---- fragment loaders (all 16B-aligned vector LDS/global loads) ------------
// A-matrix 16x32 f16 (ISA 7.12.2): lane<16 -> row M=lane, K={0..7,16..23};
// lane>=16 -> row M=lane-16, K={8..15,24..31}. Two contiguous 8-half chunks.
static __device__ __forceinline__ v16h afrag_load(const h16* p, int stride) {
  const int lane = threadIdx.x & 31;
  const int hi   = lane >> 4;
  const int m    = lane & 15;
  const h16* r = p + m * stride + hi * 8;
  union { float4 f; h16 h[8]; } u0, u1;
  u0.f = *(const float4*)(r);
  u1.f = *(const float4*)(r + 16);
  v16h a;
#pragma unroll
  for (int e = 0; e < 8; ++e) { a[e] = u0.h[e]; a[e + 8] = u1.h[e]; }
  return a;
}

// B-matrix 32x16 f16: lane holds column n=lane%16; lanes<16 cover K={0..7,16..23},
// lanes>=16 cover K={8..15,24..31} (matches A). Source must be [n][k]
// row-major (K contiguous) => two ds_load_b128 per fragment.
static __device__ __forceinline__ v16h bfrag_load(const h16* p, int stride, int k0) {
  const int lane = threadIdx.x & 31;
  const int hi   = lane >> 4;
  const int n    = lane & 15;
  const h16* r = p + n * stride + k0 + hi * 8;
  union { float4 f; h16 h[8]; } u0, u1;
  u0.f = *(const float4*)(r);
  u1.f = *(const float4*)(r + 16);
  v16h b;
#pragma unroll
  for (int e = 0; e < 8; ++e) { b[e] = u0.h[e]; b[e + 8] = u1.h[e]; }
  return b;
}

// ---- LayerNorm: one 256-thread block per row (D = 2048), vectorized --------
__global__ void __launch_bounds__(256)
ln_kernel(const float* __restrict__ x, const float* __restrict__ g,
          const float* __restrict__ bb, h16* __restrict__ out, int Dn) {
  const int row = blockIdx.x;
  const float* xr = x + (size_t)row * Dn;
  const int tid = threadIdx.x;
  float4 v[2];
  float s = 0.f, s2 = 0.f;
#pragma unroll
  for (int it = 0; it < 2; ++it) {
    v[it] = ((const float4*)xr)[tid + it * 256];
    s  += v[it].x + v[it].y + v[it].z + v[it].w;
    s2 += v[it].x * v[it].x + v[it].y * v[it].y + v[it].z * v[it].z + v[it].w * v[it].w;
  }
#pragma unroll
  for (int off = 16; off >= 1; off >>= 1) {
    s  += __shfl_xor(s,  off, 32);
    s2 += __shfl_xor(s2, off, 32);
  }
  __shared__ float ps[8], ps2[8];
  const int wid = tid >> 5;
  if ((tid & 31) == 0) { ps[wid] = s; ps2[wid] = s2; }
  __syncthreads();
  float ts = 0.f, ts2 = 0.f;
#pragma unroll
  for (int w = 0; w < 8; ++w) { ts += ps[w]; ts2 += ps2[w]; }
  const float mean = ts / Dn;
  const float var  = ts2 / Dn - mean * mean;
  const float rstd = rsqrtf(var + 1e-5f);
#pragma unroll
  for (int it = 0; it < 2; ++it) {
    const int i4 = tid + it * 256;
    const float4 gv = ((const float4*)g)[i4];
    const float4 bv = ((const float4*)bb)[i4];
    union { h16 h[4]; float2 f; } o;
    o.h[0] = (h16)((v[it].x - mean) * rstd * gv.x + bv.x);
    o.h[1] = (h16)((v[it].y - mean) * rstd * gv.y + bv.y);
    o.h[2] = (h16)((v[it].z - mean) * rstd * gv.z + bv.z);
    o.h[3] = (h16)((v[it].w - mean) * rstd * gv.w + bv.w);
    ((float2*)(out + (size_t)row * Dn))[i4] = o.f;
  }
}

// ---- WMMA GEMM: C[M,N] = A_f16[M,K] * W_f32[K,N] + bias, fused epilogues ---
// 256 threads = 8 waves; block tile 128x128, wave tile 32x64, BK=32.
// W is converted f32->f16 while staging into a transposed [n][k] LDS tile.
// flags: 1=write f16, 2=write f32, 4=+residual(f32), 8=up*silu(C).
#define BM 128
#define BN 128
#define BK 32
#define BSTRIDE 40  // halves; 80B rows keep 16B alignment + spread banks
__global__ void __launch_bounds__(256)
gemm_kernel(const h16* __restrict__ A, const float* __restrict__ W,
            const float* __restrict__ bias, const float* __restrict__ residual,
            const h16* __restrict__ mulsrc, h16* __restrict__ outH,
            float* __restrict__ outF, int M, int N, int K, int flags) {
  __shared__ __align__(16) h16 As[BM * BK];          // [m][k]  8KB
  __shared__ __align__(16) h16 Bs[BN * BSTRIDE];     // [n][k] 10KB (transposed)
  const int tid = threadIdx.x;
  const int wid = tid >> 5;
  const int m0 = blockIdx.y * BM;
  const int n0 = blockIdx.x * BN;
  const int wm = (wid & 3) * 32;   // wave row offset inside block tile
  const int wn = (wid >> 2) * 64;  // wave col offset inside block tile

  const v8f vzero = {0.f, 0.f, 0.f, 0.f, 0.f, 0.f, 0.f, 0.f};
  v8f acc[2][4];
#pragma unroll
  for (int tm = 0; tm < 2; ++tm)
#pragma unroll
    for (int tn = 0; tn < 4; ++tn) acc[tm][tn] = vzero;

  const int arow = tid >> 1, aseg = tid & 1;   // A: 2 thr/row, 16 halves
  const int brow = tid >> 3, bseg = tid & 7;   // W: 8 thr/row, 16 floats

  for (int k0 = 0; k0 < K; k0 += BK) {
    // stage A tile 128x32 f16: pure copy -> async direct-to-LDS when available
    {
      const h16* src = A + (size_t)(m0 + arow) * K + k0 + aseg * 16;
      h16* dst = &As[arow * BK + aseg * 16];
#ifdef HAVE_ASYNC_LDS
      async_copy_b128(src, dst);
      async_copy_b128(src + 8, dst + 8);
#else
      float4 v0 = *(const float4*)(src);
      float4 v1 = *(const float4*)(src + 8);
      *(float4*)(dst)     = v0;
      *(float4*)(dst + 8) = v1;
#endif
    }
    // stage W tile 32x128: f32 load, f16 convert, transposed [n][k] store
    {
      const float* src = W + (size_t)(k0 + brow) * N + n0 + bseg * 16;
      float4 w0 = ((const float4*)src)[0];
      float4 w1 = ((const float4*)src)[1];
      float4 w2 = ((const float4*)src)[2];
      float4 w3 = ((const float4*)src)[3];
      float wv[16] = {w0.x, w0.y, w0.z, w0.w, w1.x, w1.y, w1.z, w1.w,
                      w2.x, w2.y, w2.z, w2.w, w3.x, w3.y, w3.z, w3.w};
      h16* dst = &Bs[(bseg * 16) * BSTRIDE + brow];
#pragma unroll
      for (int i = 0; i < 16; ++i) dst[i * BSTRIDE] = (h16)wv[i];
    }
#ifdef HAVE_ASYNC_LDS
    async_wait0();
#endif
    __syncthreads();
    if (k0 + BK < K) {  // pull next K-slice toward L2/WGP$ (global_prefetch_b8)
      __builtin_prefetch(A + (size_t)(m0 + arow) * K + k0 + BK, 0, 1);
      __builtin_prefetch(W + (size_t)(k0 + BK + brow) * N + n0, 0, 1);
    }
    const v16h af0 = afrag_load(&As[(wm +  0) * BK], BK);
    const v16h af1 = afrag_load(&As[(wm + 16) * BK], BK);
#pragma unroll
    for (int tn = 0; tn < 4; ++tn) {
      const v16h bf = bfrag_load(&Bs[(wn + tn * 16) * BSTRIDE], BSTRIDE, 0);
      acc[0][tn] = WMMA16(af0, bf, acc[0][tn]);
      acc[1][tn] = WMMA16(af1, bf, acc[1][tn]);
    }
    __syncthreads();
  }

  // epilogue: C layout => lane holds (row = r + 8*hi, col = lane%16) per VGPR r
  const int lane = tid & 31, hi = lane >> 4, cl = lane & 15;
#pragma unroll
  for (int tm = 0; tm < 2; ++tm) {
#pragma unroll
    for (int tn = 0; tn < 4; ++tn) {
      const int colg = n0 + wn + tn * 16 + cl;
      const float bcol = bias[colg];
#pragma unroll
      for (int r = 0; r < 8; ++r) {
        const int rowg = m0 + wm + tm * 16 + r + hi * 8;
        const size_t idx = (size_t)rowg * N + colg;
        float c = acc[tm][tn][r] + bcol;
        if (flags & 8) c = (float)mulsrc[idx] * (c / (1.f + __expf(-c)));
        if (flags & 4) c += residual[idx];
        if (flags & 1) outH[idx] = (h16)c;
        if (flags & 2) outF[idx] = c;
      }
    }
  }
}

// ---- Flash attention with ALiBi + causal, all-WMMA -------------------------
// Grid (S/64, H, B); 128 threads = 4 waves; each wave owns a 16-row Q tile and
// streams 32-key blocks through shared K/V tiles with online softmax.
// K tile is [key][hd] (scores operand, async-copied); V tile is transposed
// [hd][key] so the PV operand also loads as ds_load_b128.
__global__ void __launch_bounds__(128)
attn_kernel(const h16* __restrict__ qkv, h16* __restrict__ ctx) {
  const int S = 2048, D = 2048, H3 = 6144, HD = 128;
  __shared__ __align__(16) h16 Ks[32 * 128];       //  8KB [key][hd]
  __shared__ __align__(16) h16 Vs[128 * BSTRIDE];  // 10KB [hd][key] transposed
  __shared__ __align__(16) h16 Pst[4][16 * 32];    //  4KB per-wave P staging

  const int qblk = blockIdx.x, h = blockIdx.y, b = blockIdx.z;
  const int tid = threadIdx.x;
  const int wid = tid >> 5, lane = tid & 31, hi = lane >> 4, cl = lane & 15;
  const int q0 = qblk * 64 + wid * 16;  // this wave's first query row
  const size_t tokBase = (size_t)b * S;

  // Q fragments for full head dim (4 x K=32 slices), kept in VGPRs
  v16h qf[4];
#pragma unroll
  for (int kc = 0; kc < 4; ++kc)
    qf[kc] = afrag_load(qkv + (tokBase + q0) * H3 + h * HD + kc * 32, H3);

  const v8f vzero = {0.f, 0.f, 0.f, 0.f, 0.f, 0.f, 0.f, 0.f};
  v8f cacc[8];
#pragma unroll
  for (int t = 0; t < 8; ++t) cacc[t] = vzero;
  float mrow[8], lrow[8];
#pragma unroll
  for (int r = 0; r < 8; ++r) { mrow[r] = -3e38f; lrow[r] = 0.f; }

  const float slope  = exp2f(-0.5f * (float)(h + 1));  // 2^(-8*(h+1)/H)
  const float inv_hd = 1.0f / 128.0f;                  // muP: scores / head_dim
  const int jmax = (qblk * 64 + 63) >> 5;              // causal key-block bound

  const int krow = tid >> 2, kquad = tid & 3;  // K/V staging: 4 thr/key row

  for (int j = 0; j <= jmax; ++j) {
    const int key0 = j * 32;
    // stage K [key][hd] (pure copy -> async) and V transposed [hd][key]
    {
      const h16* ksrc = qkv + (tokBase + key0 + krow) * H3 + D + h * HD + kquad * 32;
      const h16* vsrc = ksrc + D;
      h16* kd = &Ks[krow * 128 + kquad * 32];
#ifdef HAVE_ASYNC_LDS
#pragma unroll
      for (int i = 0; i < 4; ++i) async_copy_b128(ksrc + i * 8, kd + i * 8);
#else
#pragma unroll
      for (int i = 0; i < 4; ++i) *(float4*)(kd + i * 8) = *(const float4*)(ksrc + i * 8);
#endif
#pragma unroll
      for (int i = 0; i < 4; ++i) {
        union { float4 f; h16 h[8]; } u;
        u.f = *(const float4*)(vsrc + i * 8);
#pragma unroll
        for (int e = 0; e < 8; ++e)
          Vs[(kquad * 32 + i * 8 + e) * BSTRIDE + krow] = u.h[e];
      }
    }
#ifdef HAVE_ASYNC_LDS
    async_wait0();
#endif
    __syncthreads();

    // scores: two 16-key tiles, K-depth 128 via 4 WMMAs each
    v8f s[2] = {vzero, vzero};
#pragma unroll
    for (int kc = 0; kc < 4; ++kc) {
      const v16h b0 = bfrag_load(Ks,            128, kc * 32);
      const v16h b1 = bfrag_load(Ks + 16 * 128, 128, kc * 32);
      s[0] = WMMA16(qf[kc], b0, s[0]);
      s[1] = WMMA16(qf[kc], b1, s[1]);
    }

    // scale + ALiBi + causal mask + online softmax update
    float scale[8];
#pragma unroll
    for (int r = 0; r < 8; ++r) {
      const int rowabs = q0 + r + hi * 8;
      float mx = mrow[r];
#pragma unroll
      for (int t = 0; t < 2; ++t) {
        const int key = key0 + t * 16 + cl;
        float v = s[t][r] * inv_hd + slope * (float)(key - rowabs);
        v = (key <= rowabs) ? v : -3e38f;
        s[t][r] = v;
        mx = fmaxf(mx, v);
      }
#pragma unroll
      for (int off = 1; off < 16; off <<= 1) mx = fmaxf(mx, __shfl_xor(mx, off, 32));
      scale[r] = __expf(mrow[r] - mx);
      mrow[r] = mx;
      float rs = 0.f;
#pragma unroll
      for (int t = 0; t < 2; ++t) {
        const float p = __expf(s[t][r] - mx);
        s[t][r] = p;
        rs += p;
      }
#pragma unroll
      for (int off = 1; off < 16; off <<= 1) rs += __shfl_xor(rs, off, 32);
      lrow[r] = lrow[r] * scale[r] + rs;
    }
#pragma unroll
    for (int t = 0; t < 8; ++t)
#pragma unroll
      for (int r = 0; r < 8; ++r) cacc[t][r] *= scale[r];

    // stage P (C-layout -> A-layout) through this wave's private LDS tile
    h16* pw = &Pst[wid][0];
#pragma unroll
    for (int t = 0; t < 2; ++t)
#pragma unroll
      for (int r = 0; r < 8; ++r)
        pw[(r + hi * 8) * 32 + t * 16 + cl] = (h16)s[t][r];
    const v16h pf = afrag_load(pw, 32);

    // ctx += P(16x32) * V(32x128): 8 WMMAs across head dim
#pragma unroll
    for (int tn = 0; tn < 8; ++tn) {
      const v16h bv = bfrag_load(&Vs[(tn * 16) * BSTRIDE], BSTRIDE, 0);
      cacc[tn] = WMMA16(pf, bv, cacc[tn]);
    }
    __syncthreads();
  }

  // normalize and write ctx in [B,S,H*HD] layout (matches heads-merge reshape)
#pragma unroll
  for (int tn = 0; tn < 8; ++tn)
#pragma unroll
    for (int r = 0; r < 8; ++r) {
      const int rowabs = q0 + r + hi * 8;
      const float v = cacc[tn][r] / lrow[r];
      ctx[(tokBase + rowabs) * (size_t)D + h * HD + tn * 16 + cl] = (h16)v;
    }
}

// ============================================================================
extern "C" void kernel_launch(void* const* d_in, const int* in_sizes, int n_in,
                              void* d_out, int out_size, void* d_ws, size_t ws_size,
                              hipStream_t stream) {
  (void)in_sizes; (void)n_in; (void)out_size; (void)ws_size;
  const int B = 2, S = 2048, D = 2048, H = 16, F = 6144;
  const int M = B * S;

  const float* hidden = (const float*)d_in[0];
  // d_in[1] attention_mask / d_in[2] position_bias: recomputed analytically
  const float* ln1_g  = (const float*)d_in[3];
  const float* ln1_b  = (const float*)d_in[4];
  const float* w_qkv  = (const float*)d_in[5];
  const float* b_qkv  = (const float*)d_in[6];
  const float* w_o    = (const float*)d_in[7];
  const float* b_o    = (const float*)d_in[8];
  const float* ln2_g  = (const float*)d_in[9];
  const float* ln2_b  = (const float*)d_in[10];
  const float* w_fc   = (const float*)d_in[11];
  const float* b_fc   = (const float*)d_in[12];
  const float* w_fc2  = (const float*)d_in[13];
  const float* b_fc2  = (const float*)d_in[14];
  const float* w_proj = (const float*)d_in[15];
  const float* b_proj = (const float*)d_in[16];

  // workspace layout (bump allocator, 256B aligned); ~168MB total
  char* ws = (char*)d_ws;
  size_t off = 0;
  auto alloc = [&](size_t bytes) {
    size_t o = off;
    off += (bytes + 255) & ~(size_t)255;
    return o;
  };
  h16*   hF16   = (h16*)(ws + alloc((size_t)M * D * sizeof(h16)));     // LN out (reused)
  h16*   qkvF16 = (h16*)(ws + alloc((size_t)M * 3 * D * sizeof(h16))); // qkv, later 'up'
  h16*   ctxF16 = (h16*)(ws + alloc((size_t)M * D * sizeof(h16)));
  float* hidF   = (float*)(ws + alloc((size_t)M * D * sizeof(float))); // post-attn hidden
  h16*   actF16 = (h16*)(ws + alloc((size_t)M * F * sizeof(h16)));
  h16*   upF16  = qkvF16;

  // 1. LN1 -> f16
  ln_kernel<<<M, 256, 0, stream>>>(hidden, ln1_g, ln1_b, hF16, D);
  // 2. QKV projection (weight converted in-kernel)
  gemm_kernel<<<dim3(3 * D / BN, M / BM), 256, 0, stream>>>(
      hF16, w_qkv, b_qkv, nullptr, nullptr, qkvF16, nullptr, M, 3 * D, D, 1);
  // 3. flash attention (causal + ALiBi inline)
  attn_kernel<<<dim3(S / 64, H, B), 128, 0, stream>>>(qkvF16, ctxF16);
  // 4. output projection + bias + residual -> f32 hidden
  gemm_kernel<<<dim3(D / BN, M / BM), 256, 0, stream>>>(
      ctxF16, w_o, b_o, hidden, nullptr, nullptr, hidF, M, D, D, 6);
  // 5. LN2 -> f16
  ln_kernel<<<M, 256, 0, stream>>>(hidF, ln2_g, ln2_b, hF16, D);
  // 6. up = h2 @ w_fc + b_fc
  gemm_kernel<<<dim3(F / BN, M / BM), 256, 0, stream>>>(
      hF16, w_fc, b_fc, nullptr, nullptr, upF16, nullptr, M, F, D, 1);
  // 7. act = up * silu(h2 @ w_fc2 + b_fc2)
  gemm_kernel<<<dim3(F / BN, M / BM), 256, 0, stream>>>(
      hF16, w_fc2, b_fc2, nullptr, upF16, actF16, nullptr, M, F, D, 9);
  // 8. down projection + bias + residual -> d_out (f32)
  gemm_kernel<<<dim3(D / BN, M / BM), 256, 0, stream>>>(
      actF16, w_proj, b_proj, hidF, nullptr, nullptr, (float*)d_out, M, D, F, 6);
}
